// Encoder_22247930593892
// MI455X (gfx1250) — matrix operand order repaired and checked by reading, not compiled, not run on hardware
//
#include <hip/hip_runtime.h>
#include <hip/hip_bf16.h>

// ---------- CDNA5 vector types (match probe-confirmed WMMA builtin sigs) ----
typedef __attribute__((ext_vector_type(16))) __bf16 bf16x16;
typedef __attribute__((ext_vector_type(8)))  __bf16 bf16x8;
typedef __attribute__((ext_vector_type(8)))  float  f32x8;

static constexpr int Bsz = 512;   // batch
static constexpr int Tt  = 128;   // time steps
static constexpr int Nn  = 256;   // feature dim
static constexpr int Mm  = 128;   // hidden dim
static constexpr int KG  = 384;   // gates K = N + M
static constexpr int Gg  = 512;   // 4*M gates
static constexpr int BPG = 8;     // batch rows per workgroup

// ---------- fast math on gfx1250 -------------------------------------------
__device__ __forceinline__ float fast_tanh(float x) {
#if __has_builtin(__builtin_amdgcn_tanhf)
  return __builtin_amdgcn_tanhf(x);
#elif __has_builtin(__builtin_amdgcn_tanh_f32)
  return __builtin_amdgcn_tanh_f32(x);
#else
  float y;
  asm("v_tanh_f32 %0, %1" : "=v"(y) : "v"(x));
  return y;
#endif
}
__device__ __forceinline__ float sigm(float x) {
  // 1 / (1 + 2^(-x*log2e))  -> v_exp_f32 + v_rcp_f32
  return __builtin_amdgcn_rcpf(1.0f + __builtin_amdgcn_exp2f(-1.44269504f * x));
}
__device__ __forceinline__ float wred_max(float v) {
#pragma unroll
  for (int o = 16; o > 0; o >>= 1) v = fmaxf(v, __shfl_xor(v, o, 32));
  return v;
}
__device__ __forceinline__ float wred_sum(float v) {
#pragma unroll
  for (int o = 16; o > 0; o >>= 1) v += __shfl_xor(v, o, 32);
  return v;
}

// ---------- WMMA fragment loaders ------------------------------------------
// A (16xK bf16) per-lane layout per ISA 7.12.2: lane<16 -> row M=lane, K chunks
// {8*(lane/16) .. +7} and {16+8*(lane/16) .. +7} within each 32-wide K block.
__device__ __forceinline__ bf16x16 ldsAfrag(const __bf16* base, int stride,
                                            int lane, int kb) {
  const int row = lane & 15;
  const int k0  = kb * 32 + ((lane >> 4) << 3);
  const __bf16* p = base + row * stride + k0;
  bf16x8 lo = *(const bf16x8*)p;
  bf16x8 hi = *(const bf16x8*)(p + 16);
  bf16x16 a;
#pragma unroll
  for (int j = 0; j < 8; ++j) { a[j] = lo[j]; a[j + 8] = hi[j]; }
  return a;
}
// B (KxN bf16) stored N-major (row n, K contiguous): lane holds one column's
// K chunks {8*(lane/16) .. +7} and {16+8*(lane/16) .. +7} -> two 16B loads.
__device__ __forceinline__ bf16x16 gBfrag(const __bf16* Bm, int Kstride,
                                          int colbase, int lane, int kb) {
  const int col = colbase + (lane & 15);
  const int k0  = kb * 32 + ((lane >> 4) << 3);
  const __bf16* p = Bm + (size_t)col * Kstride + k0;
  bf16x8 lo = *(const bf16x8*)p;
  bf16x8 hi = *(const bf16x8*)(p + 16);
  bf16x16 b;
#pragma unroll
  for (int j = 0; j < 8; ++j) { b[j] = lo[j]; b[j + 8] = hi[j]; }
  return b;
}

// ---------- kernel 1: pack weights to bf16 (L2-resident) -------------------
__global__ void pack_weights(const float* __restrict__ We,
                             const float* __restrict__ Wih,
                             const float* __restrict__ Whh,
                             const float* __restrict__ bih,
                             const float* __restrict__ bhh,
                             __bf16* __restrict__ Whs,   // [128][256] s-major
                             __bf16* __restrict__ Wx,    // [128][128] s-major
                             __bf16* __restrict__ Wcat,  // [512][384] g-major
                             float* __restrict__ bias) { // [512] b_ih+b_hh
  const int total = 128 * 256 + 128 * 128 + 512 * 384 + 512;
  for (int i = blockIdx.x * blockDim.x + threadIdx.x; i < total;
       i += gridDim.x * blockDim.x) {
    int j = i;
    if (j < 128 * 256) {                      // W_hs = W_e[:, :256]
      int s = j >> 8, k = j & 255;
      Whs[j] = (__bf16)We[s * 384 + k];
      continue;
    }
    j -= 128 * 256;
    if (j < 128 * 128) {                      // W_x = W_e[:, 256:]
      int s = j >> 7, t = j & 127;
      Wx[j] = (__bf16)We[s * 384 + 256 + t];
      continue;
    }
    j -= 128 * 128;
    if (j < 512 * 384) {                      // [W_ih | W_hh] concat along K
      int g = j / 384, k = j % 384;
      Wcat[j] = (__bf16)(k < 256 ? Wih[g * 256 + k] : Whh[g * 128 + (k - 256)]);
      continue;
    }
    j -= 512 * 384;
    bias[j] = bih[j] + bhh[j];
  }
}

// ---------- kernel 2: feat_proj GEMM via WMMA ------------------------------
// FP[(b*256+n)][s] = sum_t X[b,t,n] * Wx[s][t];  rows=B*N=131072, K=T=128.
__global__ __launch_bounds__(256, 2)
void feat_proj_kernel(const float* __restrict__ X,
                      const __bf16* __restrict__ Wx,
                      __bf16* __restrict__ FP) {
  const int lane = threadIdx.x & 31;
  const int wv   = threadIdx.x >> 5;
  const int row0 = (blockIdx.x * 8 + wv) * 16;

  f32x8 acc[8] = {};
  const int arow = row0 + (lane & 15);
  // A element (row,t) = X[b*T*N + t*N + n]; adjacent lanes -> adjacent n.
  const float* xb = X + (size_t)(arow >> 8) * (Tt * Nn) + (arow & 255);
#pragma unroll
  for (int kb = 0; kb < 4; ++kb) {
    bf16x16 a;
    const int t0 = kb * 32 + ((lane >> 4) << 3);
#pragma unroll
    for (int j = 0; j < 8; ++j) {
      a[j]     = (__bf16)xb[(size_t)(t0 + j) * Nn];
      a[j + 8] = (__bf16)xb[(size_t)(t0 + 16 + j) * Nn];
    }
#pragma unroll
    for (int ct = 0; ct < 8; ++ct) {
      bf16x16 b = gBfrag(Wx, 128, ct * 16, lane, kb);
      acc[ct] = __builtin_amdgcn_wmma_f32_16x16x32_bf16(
          false, a, false, b, (short)0, acc[ct], false, false);
    }
  }
  const int halfrow = (lane >> 4) << 3;
  const int c       = lane & 15;
#pragma unroll
  for (int ct = 0; ct < 8; ++ct) {
#pragma unroll
    for (int r = 0; r < 8; ++r) {
      const int row = row0 + r + halfrow;
      FP[(size_t)row * 128 + ct * 16 + c] = (__bf16)acc[ct][r];
    }
  }
}

// ---------- kernel 3: persistent attention-LSTM scan -----------------------
// 64 WGs x 8 batch rows; the whole T=128 scan runs locally (batches are
// independent).  LDS layout (~60 KB): state + A-matrices + reused scratch.
__global__ __launch_bounds__(512, 1)
void lstm_attn_kernel(const float* __restrict__ X,
                      const __bf16* __restrict__ FP,
                      const __bf16* __restrict__ Whs,
                      const __bf16* __restrict__ Wcat,
                      const float* __restrict__ vin,
                      const float* __restrict__ biasg,
                      float* __restrict__ out) {
  __shared__ __attribute__((aligned(16))) __bf16 hsA[16 * 256];  // [h|c] bf16 A
  __shared__ __attribute__((aligned(16))) __bf16 gA[16 * 384];   // [x~|h] bf16 A
  __shared__ __attribute__((aligned(16))) float  cst[8 * 128];   // c state fp32
  __shared__ __attribute__((aligned(16))) float  vv[128];        // v_e
  __shared__ __attribute__((aligned(16))) float  bias_s[512];
  // scratch overlays: phase A -> u[0..1023], E[1024..3071]; phase B -> gbuf[0..8191]
  __shared__ __attribute__((aligned(16))) float  scratch[8192];

  const int tid  = threadIdx.x;
  const int lane = tid & 31;
  const int wv   = tid >> 5;
  const int bg0  = blockIdx.x * BPG;

  for (int i = tid; i < 16 * 256; i += 512) hsA[i] = (__bf16)0.0f;
  for (int i = tid; i < 16 * 384; i += 512) gA[i]  = (__bf16)0.0f;
  for (int i = tid; i < 8 * 128;  i += 512) cst[i] = 0.0f;
  if (tid < 128) vv[tid] = vin[tid];
  bias_s[tid] = biasg[tid];
  __syncthreads();

  for (int t = 0; t < Tt; ++t) {
    // ---- Stage 1: u = [h,c] @ W_hs^T  (16x128, K=256) via WMMA ----
    if (wv < 8) {
      f32x8 acc = {};
#pragma unroll
      for (int kb = 0; kb < 8; ++kb) {
        bf16x16 a = ldsAfrag(hsA, 256, lane, kb);
        bf16x16 b = gBfrag(Whs, 256, wv * 16, lane, kb);
        acc = __builtin_amdgcn_wmma_f32_16x16x32_bf16(
            false, a, false, b, (short)0, acc, false, false);
      }
      const int halfrow = (lane >> 4) << 3;
#pragma unroll
      for (int r = 0; r < 8; ++r) {
        const int row = r + halfrow;
        if (row < 8) scratch[row * 128 + wv * 16 + (lane & 15)] = acc[r];
      }
    }
    __syncthreads();

    // ---- Stage 2: E[b,n] = sum_s tanh(u[b,s]+fp[b,n,s]) * v[s] ----
#pragma unroll
    for (int i = 0; i < 4; ++i) {
      const int r = tid + 512 * i;            // 2048 (b,n) rows
      const int b = r >> 8, n = r & 255;
      const __bf16* fr = FP + (((size_t)(bg0 + b)) * 256 + n) * 128;
      const float* ub  = scratch + b * 128;
      float acc = 0.f;
#pragma unroll 4
      for (int s = 0; s < 128; s += 8) {
        bf16x8 f8 = *(const bf16x8*)(fr + s);
        float4 u0 = *(const float4*)(ub + s);
        float4 u1 = *(const float4*)(ub + s + 4);
        float4 w0 = *(const float4*)(vv + s);
        float4 w1 = *(const float4*)(vv + s + 4);
        acc += fast_tanh(u0.x + (float)f8[0]) * w0.x;
        acc += fast_tanh(u0.y + (float)f8[1]) * w0.y;
        acc += fast_tanh(u0.z + (float)f8[2]) * w0.z;
        acc += fast_tanh(u0.w + (float)f8[3]) * w0.w;
        acc += fast_tanh(u1.x + (float)f8[4]) * w1.x;
        acc += fast_tanh(u1.y + (float)f8[5]) * w1.y;
        acc += fast_tanh(u1.z + (float)f8[6]) * w1.z;
        acc += fast_tanh(u1.w + (float)f8[7]) * w1.w;
      }
      scratch[1024 + r] = acc;
    }
    __syncthreads();

    // ---- Stage 3: softmax over n, build x_tilde A-matrix ----
    if (wv < 8) {
      const int b = wv;
      const float* Eb = scratch + 1024 + b * 256;
      float vals[8];
      float m = -3.0e38f;
#pragma unroll
      for (int j = 0; j < 8; ++j) {
        vals[j] = Eb[lane + 32 * j];
        m = fmaxf(m, vals[j]);
      }
      m = wred_max(m);
      float ssum = 0.f;
#pragma unroll
      for (int j = 0; j < 8; ++j) {
        vals[j] = __builtin_amdgcn_exp2f((vals[j] - m) * 1.44269504f);
        ssum += vals[j];
      }
      ssum = wred_sum(ssum);
      const float inv = __builtin_amdgcn_rcpf(ssum);
      const float* xr = X + ((size_t)(bg0 + b)) * (Tt * Nn) + (size_t)t * Nn;
      if (t + 1 < Tt) __builtin_prefetch(xr + Nn, 0, 0);  // next step's x_t
#pragma unroll
      for (int j = 0; j < 8; ++j) {
        const int n = lane + 32 * j;
        gA[b * 384 + n] = (__bf16)(xr[n] * vals[j] * inv);
      }
    }
    __syncthreads();

    // ---- Stage 4: gates = [x~,h] @ [W_ih|W_hh]^T (16x512, K=384) ----
    {
      f32x8 acc0 = {}, acc1 = {};
      const int nt0 = wv * 2, nt1 = wv * 2 + 1;  // 32 col tiles / 16 waves
#pragma unroll
      for (int kb = 0; kb < 12; ++kb) {
        bf16x16 a  = ldsAfrag(gA, 384, lane, kb);
        bf16x16 b0 = gBfrag(Wcat, KG, nt0 * 16, lane, kb);
        acc0 = __builtin_amdgcn_wmma_f32_16x16x32_bf16(
            false, a, false, b0, (short)0, acc0, false, false);
        bf16x16 b1 = gBfrag(Wcat, KG, nt1 * 16, lane, kb);
        acc1 = __builtin_amdgcn_wmma_f32_16x16x32_bf16(
            false, a, false, b1, (short)0, acc1, false, false);
      }
      const int halfrow = (lane >> 4) << 3;
      const int c = lane & 15;
#pragma unroll
      for (int r = 0; r < 8; ++r) {
        const int row = r + halfrow;
        scratch[row * 512 + nt0 * 16 + c] = acc0[r];
        scratch[row * 512 + nt1 * 16 + c] = acc1[r];
      }
    }
    __syncthreads();

    // ---- Stage 5: pointwise LSTM update + write h ----
#pragma unroll
    for (int i = 0; i < 2; ++i) {
      const int p = tid + 512 * i;            // 1024 (b,m) pairs
      const int b = p >> 7, m = p & 127;
      const float* gb = scratch + b * 512;
      const float ig = sigm(gb[m]           + bias_s[m]);
      const float fg = sigm(gb[128 + m]     + bias_s[128 + m]);
      const float gg = fast_tanh(gb[256 + m] + bias_s[256 + m]);
      const float og = sigm(gb[384 + m]     + bias_s[384 + m]);
      const float c2 = fg * cst[p] + ig * gg;
      const float h2 = og * fast_tanh(c2);
      cst[p] = c2;
      hsA[b * 256 + m]       = (__bf16)h2;   // h for next u-GEMM
      hsA[b * 256 + 128 + m] = (__bf16)c2;   // c for next u-GEMM
      gA[b * 384 + 256 + m]  = (__bf16)h2;   // h for next gates-GEMM
      out[(size_t)t * (Bsz * Mm) + (size_t)(bg0 + b) * Mm + m] = h2;
    }
    __syncthreads();
  }
}

// ---------- host launcher ---------------------------------------------------
extern "C" void kernel_launch(void* const* d_in, const int* in_sizes, int n_in,
                              void* d_out, int out_size, void* d_ws,
                              size_t ws_size, hipStream_t stream) {
  const float* X   = (const float*)d_in[0];
  const float* We  = (const float*)d_in[1];
  const float* ve  = (const float*)d_in[2];
  const float* Wih = (const float*)d_in[3];
  const float* Whh = (const float*)d_in[4];
  const float* bih = (const float*)d_in[5];
  const float* bhh = (const float*)d_in[6];

  char* ws = (char*)d_ws;
  const size_t szFP   = (size_t)Bsz * Nn * Tt * 2;  // 33.55 MB bf16
  const size_t szWhs  = 128 * 256 * 2;
  const size_t szWx   = 128 * 128 * 2;
  const size_t szWcat = 512 * 384 * 2;
  __bf16* FP   = (__bf16*)(ws);
  __bf16* Whs  = (__bf16*)(ws + szFP);
  __bf16* Wx   = (__bf16*)(ws + szFP + szWhs);
  __bf16* Wcat = (__bf16*)(ws + szFP + szWhs + szWx);
  float*  bias = (float*) (ws + szFP + szWhs + szWx + szWcat);

  pack_weights<<<512, 256, 0, stream>>>(We, Wih, Whh, bih, bhh,
                                        Whs, Wx, Wcat, bias);
  feat_proj_kernel<<<1024, 256, 0, stream>>>(X, Wx, FP);
  lstm_attn_kernel<<<Bsz / BPG, 512, 0, stream>>>(X, FP, Whs, Wcat, ve, bias,
                                                  (float*)d_out);
}
